// CrossAttention_68487548502272
// MI455X (gfx1250) — matrix operand order repaired and checked by reading, not compiled, not run on hardware
//
#include <hip/hip_runtime.h>

#define BDIM  4
#define CDIM  256
#define NDIM  4096
#define MDIM  4096
#define RCDIM 32
#define WAVES 4

typedef __attribute__((ext_vector_type(16))) _Float16 v16h;
typedef __attribute__((ext_vector_type(8)))  _Float16 v8h;
typedef __attribute__((ext_vector_type(2)))  _Float16 v2h;
typedef __attribute__((ext_vector_type(8)))  float    v8f;
typedef __attribute__((ext_vector_type(4)))  float    v4f;
typedef __attribute__((ext_vector_type(8)))  unsigned int v8u;

// ---------------------------------------------------------------------------
// Projection GEMM: out[r][m] = (sum_c W[r][c] * in[b][c][m] + bias[r])*outscale
// f32 math, f16 output. NMAJOR=true  -> out[(b*M+m)*Rtot + r]  (Q/K layout)
//                       NMAJOR=false -> out[(b*Rtot+r)*M + m]  (V layout)
// ---------------------------------------------------------------------------
template<bool NMAJOR>
__global__ __launch_bounds__(256) void proj_kernel(
    const float* __restrict__ W, const float* __restrict__ bias,
    const float* __restrict__ in, _Float16* __restrict__ out, int Rtot,
    float outscale)
{
  __shared__ float Ws[32][33];
  __shared__ float Xs[32][65];
  const int m0  = blockIdx.x * 64;
  const int r0  = blockIdx.y * 32;
  const int b   = blockIdx.z;
  const int tid = threadIdx.x;

  const int mm = tid & 63;   // 0..63
  const int rr = tid >> 6;   // 0..3

  float acc[8];
  #pragma unroll
  for (int i = 0; i < 8; ++i) acc[i] = 0.f;

  for (int c0 = 0; c0 < CDIM; c0 += 32) {
    {
      int j = tid & 31, i = tid >> 5;                 // i: 0..7
      #pragma unroll
      for (int s = 0; s < 4; ++s)
        Ws[i + 8 * s][j] = W[(size_t)(r0 + i + 8 * s) * CDIM + (c0 + j)];
    }
    {
      int jm = tid & 63, ic = tid >> 6;               // ic: 0..3
      #pragma unroll
      for (int s = 0; s < 8; ++s)
        Xs[ic + 4 * s][jm] =
            in[((size_t)b * CDIM + (c0 + ic + 4 * s)) * MDIM + (m0 + jm)];
    }
    __syncthreads();
    #pragma unroll
    for (int cc = 0; cc < 32; ++cc) {
      float xv = Xs[cc][mm];
      #pragma unroll
      for (int i = 0; i < 8; ++i)
        acc[i] += Ws[rr + 4 * i][cc] * xv;
    }
    __syncthreads();
  }

  #pragma unroll
  for (int i = 0; i < 8; ++i) {
    int r = r0 + rr + 4 * i;
    float v = (acc[i] + bias[r]) * outscale;
    if (NMAJOR)
      out[((size_t)b * MDIM + (m0 + mm)) * Rtot + r] = (_Float16)v;
    else
      out[((size_t)b * Rtot + r) * MDIM + (m0 + mm)] = (_Float16)v;
  }
}

// ---------------------------------------------------------------------------
// Flash cross-attention, reduction-free softmax (logits bounded -> no max
// subtraction needed; row-sum computed by an extra WMMA against a ones
// matrix, accumulated in a persistent D fragment).
// One wave owns 16 query rows; streams M in 32-wide tiles:
//   2x v_wmma (S = Q K^T tile), exp, P->LDS (packed dwords) -> A fragment,
//   1x v_wmma (row-sum accum), 16x v_wmma (O += P V^T per 16-d tile).
// Q: (B,N,32) f16 (pre-scaled by 1/sqrt(32)); K: (B,M,32) f16; V: (B,C,M) f16.
// ---------------------------------------------------------------------------
__global__ __launch_bounds__(32 * WAVES) void flashattn_kernel(
    const _Float16* __restrict__ Qh, const _Float16* __restrict__ Kh,
    const _Float16* __restrict__ Vh, const float* __restrict__ x,
    const float* __restrict__ gamma, float* __restrict__ out)
{
  // plds32[w][row][wd] packs (P[row][wd], P[row][wd+16]) in one dword
  __shared__ __align__(32) unsigned int plds32[WAVES][16][16];

  const int b    = blockIdx.y;
  const int wave = threadIdx.x >> 5;
  const int lane = threadIdx.x & 31;
  const int half = lane >> 4;   // 0/1 lane half
  const int lid  = lane & 15;   // 0..15
  const int n0   = blockIdx.x * (WAVES * 16) + wave * 16;

  // Q A-fragment: lane row = lid; elems 0..15 <- K[16h..16h+15]
  const _Float16* qbase = Qh + ((size_t)b * NDIM + (n0 + lid)) * RCDIM;
  v8h q0 = *(const v8h*)(qbase + 16 * half);
  v8h q1 = *(const v8h*)(qbase + 16 * half + 8);
  v16h a_q = __builtin_shufflevector(q0, q1, 0, 1, 2, 3, 4, 5, 6, 7,
                                     8, 9, 10, 11, 12, 13, 14, 15);

  v16h ones;
  #pragma unroll
  for (int e = 0; e < 16; ++e) ones[e] = (_Float16)1.0f;

  v8f zero = {};
  v8f rs = zero;          // running row-sum fragment (all 16 cols identical)
  v8f o[16];
  #pragma unroll
  for (int t = 0; t < 16; ++t) o[t] = zero;

  const _Float16* kb = Kh + (size_t)b * MDIM * RCDIM;
  const _Float16* vb = Vh + (size_t)b * CDIM * MDIM;

  #pragma unroll 1
  for (int m0 = 0; m0 < MDIM; m0 += 32) {
    // K B-fragments: lane col = lid, 16 contiguous K values starting at 16h
    v16h bk0 = *(const v16h*)(kb + (size_t)(m0 + lid) * RCDIM + 16 * half);
    v16h bk1 = *(const v16h*)(kb + (size_t)(m0 + 16 + lid) * RCDIM + 16 * half);

    v8f s0 = __builtin_amdgcn_wmma_f32_16x16x32_f16(false, a_q, false, bk0,
                                                    (short)0, zero, false, false);
    v8f s1 = __builtin_amdgcn_wmma_f32_16x16x32_f16(false, a_q, false, bk1,
                                                    (short)0, zero, false, false);

    // P = exp(S): lane holds rows j+8h, cols lid (s0) / lid+16 (s1).
    // Pack the two cols of a row into one dword and store to LDS.
    #pragma unroll
    for (int j = 0; j < 8; ++j) {
      v2h pp;
      pp[0] = (_Float16)__expf(s0[j]);
      pp[1] = (_Float16)__expf(s1[j]);
      plds32[wave][j + 8 * half][lid] = __builtin_bit_cast(unsigned int, pp);
    }
    asm volatile("s_wait_dscnt 0" ::: "memory");  // per-wave LDS RAW

    // A-fragment for P: lane row = lid, 8 packed dwords starting at 8*half,
    // then deinterleave (lo halves = m 8h..8h+7, hi halves = m 16+8h..).
    v8u raw = *(const v8u*)(&plds32[wave][lid][8 * half]);
    v16h ri = __builtin_bit_cast(v16h, raw);
    v16h a_p = __builtin_shufflevector(ri, ri, 0, 2, 4, 6, 8, 10, 12, 14,
                                       1, 3, 5, 7, 9, 11, 13, 15);

    // row-sum via matrix pipe: rs += P * ones
    rs = __builtin_amdgcn_wmma_f32_16x16x32_f16(false, a_p, false, ones,
                                                (short)0, rs, false, false);

    // O(16n x 256d) += P(16x32) * V^T(32x16) per d-tile
    #pragma unroll
    for (int t = 0; t < 16; ++t) {
      v8h v0 = *(const v8h*)(vb + (size_t)(t * 16 + lid) * MDIM +
                             m0 + 8 * half);
      v8h v1 = *(const v8h*)(vb + (size_t)(t * 16 + lid) * MDIM +
                             m0 + 16 + 8 * half);
      v16h bv = __builtin_shufflevector(v0, v1, 0, 1, 2, 3, 4, 5, 6, 7,
                                        8, 9, 10, 11, 12, 13, 14, 15);
      o[t] = __builtin_amdgcn_wmma_f32_16x16x32_f16(false, a_p, false, bv,
                                                    (short)0, o[t], false, false);
    }
  }

  const float g = gamma[0];
  float inv[8];
  #pragma unroll
  for (int j = 0; j < 8; ++j) inv[j] = 1.f / rs[j];

  // Epilogue: lane's 8 rows are consecutive n for fixed d -> b128 I/O.
  #pragma unroll
  for (int t = 0; t < 16; ++t) {
    int d = t * 16 + lid;
    size_t base = ((size_t)b * CDIM + d) * NDIM + n0 + 8 * half;
    v4f x0 = *(const v4f*)(x + base);
    v4f x1 = *(const v4f*)(x + base + 4);
    v4f r0, r1;
    #pragma unroll
    for (int e = 0; e < 4; ++e) {
      r0[e] = g * (o[t][e] * inv[e]) + x0[e];
      r1[e] = g * (o[t][4 + e] * inv[4 + e]) + x1[e];
    }
    *(v4f*)(out + base)     = r0;
    *(v4f*)(out + base + 4) = r1;
  }
}

// ---------------------------------------------------------------------------
extern "C" void kernel_launch(void* const* d_in, const int* in_sizes, int n_in,
                              void* d_out, int out_size, void* d_ws, size_t ws_size,
                              hipStream_t stream) {
  (void)in_sizes; (void)n_in; (void)out_size; (void)ws_size;
  const float* x     = (const float*)d_in[0];
  const float* ctx   = (const float*)d_in[1];
  const float* Wq    = (const float*)d_in[2];
  const float* bq    = (const float*)d_in[3];
  const float* Wk    = (const float*)d_in[4];
  const float* bk    = (const float*)d_in[5];
  const float* Wv    = (const float*)d_in[6];
  const float* bv    = (const float*)d_in[7];
  const float* gamma = (const float*)d_in[8];
  float* out = (float*)d_out;

  _Float16* Qh = (_Float16*)d_ws;                       // 1 MB
  _Float16* Kh = Qh + (size_t)BDIM * NDIM * RCDIM;      // 1 MB
  _Float16* Vh = Kh + (size_t)BDIM * MDIM * RCDIM;      // 8 MB

  const float scale = 0.17677669529663687f;  // 1/sqrt(32), folded into Q

  dim3 blk(256);
  proj_kernel<true ><<<dim3(MDIM / 64, 1, BDIM), blk, 0, stream>>>(Wq, bq, x,   Qh, RCDIM, scale);
  proj_kernel<true ><<<dim3(MDIM / 64, 1, BDIM), blk, 0, stream>>>(Wk, bk, ctx, Kh, RCDIM, 1.0f);
  proj_kernel<false><<<dim3(MDIM / 64, CDIM / 32, BDIM), blk, 0, stream>>>(Wv, bv, ctx, Vh, CDIM, 1.0f);

  flashattn_kernel<<<dim3(NDIM / (WAVES * 16), BDIM), dim3(32 * WAVES), 0, stream>>>(
      Qh, Kh, Vh, x, gamma, out);
}
